// Informer_42193758716151
// MI455X (gfx1250) — matrix operand order repaired and checked by reading, not compiled, not run on hardware
//
#include <hip/hip_runtime.h>
#include <math.h>

#define Bc   4
#define Lseq 1024
#define Dm   512
#define Hn   8
#define Eh   64
#define DFF  2048
#define SK   35
#define NC   10

typedef _Float16 half_t;
typedef __attribute__((ext_vector_type(16))) _Float16 v16h;
typedef __attribute__((ext_vector_type(8)))  float    v8f;
typedef __attribute__((ext_vector_type(4)))  unsigned int u32x4;
typedef __attribute__((ext_vector_type(8)))  int         i32x8;
typedef __attribute__((ext_vector_type(4)))  int         i32x4;

#if defined(__has_builtin)
#if __has_builtin(__builtin_amdgcn_tensor_load_to_lds) && __has_builtin(__builtin_amdgcn_s_wait_tensorcnt)
#define HAVE_TDM 1
#endif
#endif
#ifndef HAVE_TDM
#define HAVE_TDM 0
#endif

#if HAVE_TDM
// Issue one TDM descriptor moving a 128-row x 32-half f16 tile from Wt[N][K]
// (row-major, rows K halfs apart) into LDS with 80B row stride (64B data +
// 16B pad via pad_interval=3 (16 DWORDs), pad_amount=3 (4 DWORDs)).
// 6-arg builtin form (clang-23 / therock-10.0 headers).
__device__ __forceinline__ void tdm_load_b_tile(const half_t* gsrc, unsigned lds_off,
                                                int Kdim, int Ndim)
{
    unsigned long long ga = (unsigned long long)(size_t)gsrc;
    u32x4 g0;
    g0[0] = 1u;                                    // count=1, user descriptor
    g0[1] = lds_off;                               // lds_addr (bits 63:32)
    g0[2] = (unsigned)(ga & 0xffffffffu);          // global_addr[31:0]
    g0[3] = (unsigned)((ga >> 32) & 0x1ffffffu)    // global_addr[56:32]
          | 0x80000000u;                           // type = 2 ("image")
    unsigned td0 = (unsigned)Kdim;                 // tensor_dim0 (halfs)
    unsigned td1 = (unsigned)Ndim;                 // tensor_dim1 (rows)
    i32x8 g1;
    g1[0] = (int)((1u << 16)                       // data_size = 2 bytes
                | (1u << 20)                       // pad_enable
                | (3u << 22)                       // pad_interval: 16 DWORDs = 64B
                | (3u << 25));                     // pad_amount: 4 DWORDs = 16B
    g1[1] = (int)((td0 & 0xffffu) << 16);          // [47:32]=atomic_barrier=0, [63:48]=td0 lo
    g1[2] = (int)((td0 >> 16) | ((td1 & 0xffffu) << 16));
    g1[3] = (int)((td1 >> 16) | (32u << 16));      // tile_dim0 = 32 halfs
    g1[4] = (int)128u;                             // tile_dim1 = 128 rows, tile_dim2 = 0
    g1[5] = (int)(unsigned)Kdim;                   // tensor_dim0_stride[31:0]
    g1[6] = 0;                                     // stride hi, dim1_stride lo
    g1[7] = 0;
    i32x4 z4 = {};
    i32x8 z8 = {};
    __builtin_amdgcn_tensor_load_to_lds(g0, g1, z4, z4, z8, 0);
}
#endif

// ---------------------------------------------------------------------------
// GEMM: C[M,N] = act(A[M,K] @ W[K,N] + bias[N]).  A f32; W supplied
// pre-transposed+converted as Wt[N][K] f16.  f16 WMMA, f32 accumulate.
// Block = 256 threads (8 waves), tile 64(M) x 128(N), BK=32.
// Wave tile 32x32 -> 4x v_wmma_f32_16x16x32_f16 per k-step.
// B tile staged by TDM (double-buffered) when available.
// ---------------------------------------------------------------------------
__global__ void __launch_bounds__(256)
gemm_wmma_kernel(const float* __restrict__ A, const half_t* __restrict__ Wt,
                 const float* __restrict__ bias, float* __restrict__ C,
                 int M, int N, int K, int apply_gelu)
{
    __shared__ half_t As[64 * 40];          // As[m][k], 40-half stride (80B)
    __shared__ half_t Bt[2][128 * 40];      // Bt[buf][n][k], 40-half stride

    const int tid  = threadIdx.x;
    const int wave = tid >> 5;
    const int lane = tid & 31;
    const int block_m = blockIdx.y * 64;
    const int block_n = blockIdx.x * 128;
    const int wm = (wave >> 2) * 32;        // 0,32
    const int wn = (wave &  3) * 32;        // 0,32,64,96

    v8f acc00 = {}, acc01 = {}, acc10 = {}, acc11 = {};

    const int a_row = tid >> 2;             // 0..63
    const int a_col = (tid & 3) * 8;        // 0,8,16,24
#if !HAVE_TDM
    const int b_row = tid >> 1;             // 0..127
    const int b_seg = (tid & 1) * 16;       // 0,16
#endif
    const int khalf = lane >> 4;
    const int l15   = lane & 15;
    const int niter = K >> 5;

#if HAVE_TDM
    const unsigned bt_off0 = (unsigned)(size_t)&Bt[0][0];
    const unsigned bt_off1 = (unsigned)(size_t)&Bt[1][0];
    if (wave == 0)
        tdm_load_b_tile(Wt + (size_t)block_n * K, bt_off0, K, N);
#endif

    for (int it = 0; it < niter; ++it) {
        const int k0 = it << 5;
        const int cur = it & 1;

        // stage A tile (f32 -> f16), all threads
        {
            const float* asrc = A + (size_t)(block_m + a_row) * K + (k0 + a_col);
            half_t* adst = &As[a_row * 40 + a_col];
#pragma unroll
            for (int j = 0; j < 8; ++j) adst[j] = (half_t)asrc[j];
        }
#if HAVE_TDM
        if (wave == 0) {
            if (it + 1 < niter)
                tdm_load_b_tile(Wt + (size_t)block_n * K + (k0 + 32),
                                (it & 1) ? bt_off0 : bt_off1, K, N);
            if (it + 1 < niter) __builtin_amdgcn_s_wait_tensorcnt(1);
            else                __builtin_amdgcn_s_wait_tensorcnt(0);
        }
#else
        // manual B stage: contiguous f16 copy (pre-transposed weights)
        {
            const half_t* bsrc = Wt + (size_t)(block_n + b_row) * K + (k0 + b_seg);
            half_t* bdst = &Bt[cur][b_row * 40 + b_seg];
#pragma unroll
            for (int j = 0; j < 16; ++j) bdst[j] = bsrc[j];
        }
#endif
        __syncthreads();

        // A fragments (rows wm, wm+16)
        v16h a0, a1, b0, b1;
        {
            const half_t* ap0 = &As[(wm + l15) * 40 + khalf * 16];
            const half_t* ap1 = &As[(wm + 16 + l15) * 40 + khalf * 16];
#pragma unroll
            for (int i = 0; i < 8; ++i) {
                a0[i] = ap0[i]; a0[i + 8] = ap0[8 + i];
                a1[i] = ap1[i]; a1[i + 8] = ap1[8 + i];
            }
        }
        // B fragments (cols wn, wn+16)
        {
            const half_t* bp0 = &Bt[cur][(wn + l15) * 40 + khalf * 16];
            const half_t* bp1 = &Bt[cur][(wn + 16 + l15) * 40 + khalf * 16];
#pragma unroll
            for (int i = 0; i < 16; ++i) { b0[i] = bp0[i]; b1[i] = bp1[i]; }
        }
        acc00 = __builtin_amdgcn_wmma_f32_16x16x32_f16(false, a0, false, b0, (short)0, acc00, false, false);
        acc01 = __builtin_amdgcn_wmma_f32_16x16x32_f16(false, a0, false, b1, (short)0, acc01, false, false);
        acc10 = __builtin_amdgcn_wmma_f32_16x16x32_f16(false, a1, false, b0, (short)0, acc10, false, false);
        acc11 = __builtin_amdgcn_wmma_f32_16x16x32_f16(false, a1, false, b1, (short)0, acc11, false, false);
        __syncthreads();
    }

    // epilogue: C layout — VGPR r: lanes 0-15 -> M=r, lanes 16-31 -> M=r+8
    const int row0 = block_m + wm + khalf * 8;
    const int row1 = row0 + 16;
    const int col0 = block_n + wn + l15;
    const int col1 = col0 + 16;
    const float bia0 = bias[col0];
    const float bia1 = bias[col1];
#pragma unroll
    for (int r = 0; r < 8; ++r) {
        size_t off0 = (size_t)(row0 + r) * N;
        size_t off1 = (size_t)(row1 + r) * N;
        float v00 = acc00[r] + bia0;
        float v01 = acc01[r] + bia1;
        float v10 = acc10[r] + bia0;
        float v11 = acc11[r] + bia1;
        if (apply_gelu) {
            v00 = 0.5f * v00 * (1.0f + erff(v00 * 0.70710678118654752f));
            v01 = 0.5f * v01 * (1.0f + erff(v01 * 0.70710678118654752f));
            v10 = 0.5f * v10 * (1.0f + erff(v10 * 0.70710678118654752f));
            v11 = 0.5f * v11 * (1.0f + erff(v11 * 0.70710678118654752f));
        }
        C[off0 + col0] = v00;
        C[off0 + col1] = v01;
        C[off1 + col0] = v10;
        C[off1 + col1] = v11;
    }
}

// ---------------------------------------------------------------------------
// Weight prep: W[K][N] f32 -> Wt[N][K] f16 (tiled transpose through LDS)
// ---------------------------------------------------------------------------
__global__ void __launch_bounds__(256)
transpose_f16_kernel(const float* __restrict__ W, half_t* __restrict__ Wt, int K, int N)
{
    __shared__ half_t tile[32][33];
    int kb = blockIdx.y * 32, nb = blockIdx.x * 32;
    int tx = threadIdx.x & 31, ty = threadIdx.x >> 5;   // 32 x 8
    for (int r = ty; r < 32; r += 8)
        tile[r][tx] = (half_t)W[(size_t)(kb + r) * N + nb + tx];
    __syncthreads();
    for (int r = ty; r < 32; r += 8)
        Wt[(size_t)(nb + r) * K + kb + tx] = tile[tx][r];
}

// ---------------------------------------------------------------------------
// Positional encoding add
// ---------------------------------------------------------------------------
__global__ void __launch_bounds__(256)
pe_add_kernel(float* __restrict__ X)
{
    int bl = blockIdx.x;
    int l  = bl & (Lseq - 1);
    for (int d = threadIdx.x; d < Dm; d += 256) {
        int i2 = d & ~1;
        float div = expf(-(float)i2 * (9.210340371976184f / (float)Dm));
        float ang = (float)l * div;
        X[(size_t)bl * Dm + d] += (d & 1) ? cosf(ang) : sinf(ang);
    }
}

// ---------------------------------------------------------------------------
// Sampled QK scores: M[b,h,i] = max_j(q.k_s) - mean_j(q.k_s)
// ---------------------------------------------------------------------------
__global__ void __launch_bounds__(256)
qk_sample_kernel(const float* __restrict__ Q, const float* __restrict__ Kb,
                 const int* __restrict__ idx, float* __restrict__ Msc)
{
    int t  = blockIdx.x * 256 + threadIdx.x;
    int i  = t & (Lseq - 1);
    int bh = t >> 10;
    int h  = bh & (Hn - 1);
    int b  = bh >> 3;
    const float* q = Q + ((size_t)(b * Lseq + i)) * Dm + h * Eh;
    float qr[Eh];
#pragma unroll
    for (int e = 0; e < Eh; ++e) qr[e] = q[e];
    float mx  = -3.4e38f;
    float sum = 0.0f;
    for (int j = 0; j < SK; ++j) {
        int s = idx[i * SK + j];
        const float* kr = Kb + ((size_t)(b * Lseq + s)) * Dm + h * Eh;
        float d = 0.0f;
#pragma unroll
        for (int e = 0; e < Eh; ++e) d += qr[e] * kr[e];
        mx = fmaxf(mx, d);
        sum += d;
    }
    Msc[t] = mx - sum * (1.0f / (float)SK);
}

// ---------------------------------------------------------------------------
// Top-k (35) per (b,h): iterative wave-level argmax (ties -> lowest index)
// ---------------------------------------------------------------------------
__global__ void topk_kernel(const float* __restrict__ Msc, int* __restrict__ idxtop)
{
    __shared__ float mloc[Lseq];
    int bh = blockIdx.x;
    int lane = threadIdx.x;   // 32 threads
    for (int i = lane; i < Lseq; i += 32) mloc[i] = Msc[(size_t)bh * Lseq + i];
    __syncthreads();
    for (int t = 0; t < SK; ++t) {
        float best = -3.4e38f;
        int bidx = Lseq;
        for (int i = lane; i < Lseq; i += 32) {
            float v = mloc[i];
            if (v > best || (v == best && i < bidx)) { best = v; bidx = i; }
        }
        for (int off = 16; off > 0; off >>= 1) {
            float ov = __shfl_xor(best, off, 32);
            int   oi = __shfl_xor(bidx, off, 32);
            if (ov > best || (ov == best && oi < bidx)) { best = ov; bidx = oi; }
        }
        if (lane == 0) { idxtop[bh * SK + t] = bidx; mloc[bidx] = -3.4e38f; }
        __syncthreads();
    }
}

// ---------------------------------------------------------------------------
// V mean over sequence
// ---------------------------------------------------------------------------
__global__ void vmean_kernel(const float* __restrict__ V, float* __restrict__ vmean)
{
    int bh = blockIdx.x;
    int e  = threadIdx.x;   // 64
    int h = bh & 7, b = bh >> 3;
    const float* base = V + (size_t)b * Lseq * Dm + h * Eh + e;
    float s = 0.0f;
    for (int l = 0; l < Lseq; ++l) s += base[(size_t)l * Dm];
    vmean[bh * Eh + e] = s * (1.0f / (float)Lseq);
}

// ---------------------------------------------------------------------------
// Full attention for the 35 selected queries per (b,h)
// ---------------------------------------------------------------------------
__global__ void __launch_bounds__(256)
attn_top_kernel(const float* __restrict__ Q, const float* __restrict__ Kb,
                const float* __restrict__ V, const int* __restrict__ idxtop,
                float* __restrict__ ctxtop)
{
    __shared__ float qs[Eh];
    __shared__ float sc[Lseq];
    __shared__ float red[256];
    int blk = blockIdx.x;
    int u  = blk % SK;
    int bh = blk / SK;
    int h = bh & 7, b = bh >> 3;
    int tid = threadIdx.x;
    int qi = idxtop[bh * SK + u];
    if (tid < Eh) qs[tid] = Q[((size_t)(b * Lseq + qi)) * Dm + h * Eh + tid];
    __syncthreads();
    for (int s = tid; s < Lseq; s += 256) {
        const float* kr = Kb + ((size_t)(b * Lseq + s)) * Dm + h * Eh;
        float d = 0.0f;
#pragma unroll
        for (int e = 0; e < Eh; ++e) d += qs[e] * kr[e];
        sc[s] = d * 0.125f;
    }
    __syncthreads();
    float lm = -3.4e38f;
    for (int s = tid; s < Lseq; s += 256) lm = fmaxf(lm, sc[s]);
    red[tid] = lm;
    __syncthreads();
    for (int off = 128; off > 0; off >>= 1) {
        if (tid < off) red[tid] = fmaxf(red[tid], red[tid + off]);
        __syncthreads();
    }
    float mx = red[0];
    __syncthreads();
    float ls = 0.0f;
    for (int s = tid; s < Lseq; s += 256) { float ev = expf(sc[s] - mx); sc[s] = ev; ls += ev; }
    red[tid] = ls;
    __syncthreads();
    for (int off = 128; off > 0; off >>= 1) {
        if (tid < off) red[tid] += red[tid + off];
        __syncthreads();
    }
    float inv = 1.0f / red[0];
    __syncthreads();
    int e = tid & 63, chunk = tid >> 6;
    float acc = 0.0f;
    const float* vb = V + (size_t)b * Lseq * Dm + h * Eh + e;
    for (int s = chunk * 256; s < chunk * 256 + 256; ++s)
        acc += sc[s] * vb[(size_t)s * Dm];
    red[tid] = acc;
    __syncthreads();
    if (tid < Eh) {
        float r = (red[tid] + red[tid + 64] + red[tid + 128] + red[tid + 192]) * inv;
        ctxtop[((size_t)(bh * SK + u)) * Eh + tid] = r;
    }
}

// ---------------------------------------------------------------------------
// Assemble attention output
// ---------------------------------------------------------------------------
__global__ void __launch_bounds__(256)
assemble_kernel(const float* __restrict__ vmean, const float* __restrict__ ctxtop,
                const int* __restrict__ idxtop, float* __restrict__ XATT)
{
    __shared__ int ufh[Hn];
    int bl = blockIdx.x;
    int l = bl & (Lseq - 1);
    int b = bl >> 10;
    int tid = threadIdx.x;
    if (tid < Hn) ufh[tid] = -1;
    __syncthreads();
    for (int t = tid; t < Hn * SK; t += 256) {
        int h = t / SK, u = t % SK;
        if (idxtop[(b * Hn + h) * SK + u] == l) ufh[h] = u;
    }
    __syncthreads();
    for (int d = tid; d < Dm; d += 256) {
        int h = d >> 6, e = d & 63;
        int u = ufh[h];
        float v = (u >= 0) ? ctxtop[((size_t)((b * Hn + h) * SK + u)) * Eh + e]
                           : vmean[(b * Hn + h) * Eh + e];
        XATT[(size_t)bl * Dm + d] = v;
    }
}

// ---------------------------------------------------------------------------
// X = LayerNorm(X [+ resid]) * g + beta   (in place on X)
// ---------------------------------------------------------------------------
__global__ void __launch_bounds__(256)
layernorm_kernel(float* __restrict__ X, const float* __restrict__ resid,
                 const float* __restrict__ g, const float* __restrict__ bta)
{
    __shared__ float row[Dm];
    __shared__ float red[256];
    int bl = blockIdx.x;
    int tid = threadIdx.x;
    size_t base = (size_t)bl * Dm;
    float s = 0.0f;
    for (int d = tid; d < Dm; d += 256) {
        float v = X[base + d];
        if (resid) v += resid[base + d];
        row[d] = v;
        s += v;
    }
    red[tid] = s;
    __syncthreads();
    for (int off = 128; off > 0; off >>= 1) { if (tid < off) red[tid] += red[tid + off]; __syncthreads(); }
    float mean = red[0] * (1.0f / (float)Dm);
    __syncthreads();
    float vs = 0.0f;
    for (int d = tid; d < Dm; d += 256) { float dv = row[d] - mean; vs += dv * dv; }
    red[tid] = vs;
    __syncthreads();
    for (int off = 128; off > 0; off >>= 1) { if (tid < off) red[tid] += red[tid + off]; __syncthreads(); }
    float rstd = rsqrtf(red[0] * (1.0f / (float)Dm) + 1e-5f);
    for (int d = tid; d < Dm; d += 256)
        X[base + d] = (row[d] - mean) * rstd * g[d] + bta[d];
}

// ---------------------------------------------------------------------------
// Mean pool + classifier
// ---------------------------------------------------------------------------
__global__ void __launch_bounds__(256)
pool_kernel(const float* __restrict__ X, float* __restrict__ pooled)
{
    int t = blockIdx.x * 256 + threadIdx.x;
    int d = t & (Dm - 1);
    int b = t >> 9;
    const float* base = X + (size_t)b * Lseq * Dm + d;
    float s = 0.0f;
    for (int l = 0; l < Lseq; ++l) s += base[(size_t)l * Dm];
    pooled[t] = s * (1.0f / (float)Lseq);
}

__global__ void classify_kernel(const float* __restrict__ pooled, const float* __restrict__ Wc,
                                const float* __restrict__ bc, float* __restrict__ out)
{
    int t = threadIdx.x;
    if (t >= Bc * NC) return;
    int b = t / NC, c = t % NC;
    float s = bc[c];
    for (int d = 0; d < Dm; ++d) s += pooled[b * Dm + d] * Wc[d * NC + c];
    out[t] = s;
}

// ---------------------------------------------------------------------------
extern "C" void kernel_launch(void* const* d_in, const int* in_sizes, int n_in,
                              void* d_out, int out_size, void* d_ws, size_t ws_size,
                              hipStream_t stream)
{
    const float* src        = (const float*)d_in[0];
    const int*   idx_sample = (const int*)  d_in[1];
    const float* Wemb  = (const float*)d_in[2];
    const float* bemb  = (const float*)d_in[3];
    const float* Wq    = (const float*)d_in[4];
    const float* bq    = (const float*)d_in[5];
    const float* Wk    = (const float*)d_in[6];
    const float* bk    = (const float*)d_in[7];
    const float* Wv    = (const float*)d_in[8];
    const float* bv    = (const float*)d_in[9];
    const float* Wo    = (const float*)d_in[10];
    const float* bo    = (const float*)d_in[11];
    const float* g1    = (const float*)d_in[12];
    const float* beta1 = (const float*)d_in[13];
    const float* W1    = (const float*)d_in[14];
    const float* bf1   = (const float*)d_in[15];
    const float* W2    = (const float*)d_in[16];
    const float* bf2   = (const float*)d_in[17];
    const float* g2    = (const float*)d_in[18];
    const float* beta2 = (const float*)d_in[19];
    const float* gf    = (const float*)d_in[20];
    const float* betaf = (const float*)d_in[21];
    const float* Wc    = (const float*)d_in[22];
    const float* bc    = (const float*)d_in[23];
    float* out = (float*)d_out;

    // workspace layout
    float* ws = (float*)d_ws;
    const size_t XN = (size_t)Bc * Lseq * Dm;          // 2,097,152 floats
    float* X    = ws;
    float* R    = X + XN;                              // QKV+XATT / FFN hidden
    float* Qb   = R;
    float* Kb   = R + XN;
    float* Vb   = R + 2 * XN;
    float* XATT = R + 3 * XN;
    float* Hb   = R;
    float* Sb   = R + 4 * XN;
    float* Msc  = Sb + XN;
    float* vmean  = Msc + (size_t)Bc * Hn * Lseq;
    float* ctxtop = vmean + Bc * Hn * Eh;
    int*   idxtop = (int*)(ctxtop + (size_t)Bc * Hn * SK * Eh);
    float* pooled = (float*)(idxtop + Bc * Hn * SK);

    // f16 transposed-weight region (halfs), 16B-aligned by construction
    half_t* wh = (half_t*)(pooled + Bc * Dm);
    half_t* WembT = wh;                                   // [Dm][32]
    size_t  hoff  = (size_t)Dm * 32;
    half_t* WqT[2]; half_t* WkT[2]; half_t* WvT[2]; half_t* WoT[2];
    half_t* W1T[2]; half_t* W2T[2];
    for (int l = 0; l < 2; ++l) {
        WqT[l] = wh + hoff; hoff += (size_t)Dm * Dm;
        WkT[l] = wh + hoff; hoff += (size_t)Dm * Dm;
        WvT[l] = wh + hoff; hoff += (size_t)Dm * Dm;
        WoT[l] = wh + hoff; hoff += (size_t)Dm * Dm;
        W1T[l] = wh + hoff; hoff += (size_t)Dm * DFF;     // [DFF][Dm]
        W2T[l] = wh + hoff; hoff += (size_t)DFF * Dm;     // [Dm][DFF]
    }

    const int Mrows = Bc * Lseq;                          // 4096
    dim3 blk(256);

    // ---- weight prep: transpose + f16 convert (deterministic, every call) ----
    transpose_f16_kernel<<<dim3(Dm / 32, 32 / 32), blk, 0, stream>>>(Wemb, WembT, 32, Dm);
    for (int l = 0; l < 2; ++l) {
        transpose_f16_kernel<<<dim3(Dm / 32, Dm / 32), blk, 0, stream>>>(Wq + (size_t)l * Dm * Dm, WqT[l], Dm, Dm);
        transpose_f16_kernel<<<dim3(Dm / 32, Dm / 32), blk, 0, stream>>>(Wk + (size_t)l * Dm * Dm, WkT[l], Dm, Dm);
        transpose_f16_kernel<<<dim3(Dm / 32, Dm / 32), blk, 0, stream>>>(Wv + (size_t)l * Dm * Dm, WvT[l], Dm, Dm);
        transpose_f16_kernel<<<dim3(Dm / 32, Dm / 32), blk, 0, stream>>>(Wo + (size_t)l * Dm * Dm, WoT[l], Dm, Dm);
        transpose_f16_kernel<<<dim3(DFF / 32, Dm / 32), blk, 0, stream>>>(W1 + (size_t)l * Dm * DFF, W1T[l], Dm, DFF);
        transpose_f16_kernel<<<dim3(Dm / 32, DFF / 32), blk, 0, stream>>>(W2 + (size_t)l * DFF * Dm, W2T[l], DFF, Dm);
    }

    // ---- embedding + positional encoding ----
    gemm_wmma_kernel<<<dim3(Dm / 128, Mrows / 64), blk, 0, stream>>>(
        src, WembT, bemb, X, Mrows, Dm, 32, 0);
    pe_add_kernel<<<Mrows, blk, 0, stream>>>(X);

    for (int layer = 0; layer < 2; ++layer) {
        const int* idx_l = idx_sample + (size_t)layer * Lseq * SK;

        gemm_wmma_kernel<<<dim3(Dm / 128, Mrows / 64), blk, 0, stream>>>(
            X, WqT[layer], bq + layer * Dm, Qb, Mrows, Dm, Dm, 0);
        gemm_wmma_kernel<<<dim3(Dm / 128, Mrows / 64), blk, 0, stream>>>(
            X, WkT[layer], bk + layer * Dm, Kb, Mrows, Dm, Dm, 0);
        gemm_wmma_kernel<<<dim3(Dm / 128, Mrows / 64), blk, 0, stream>>>(
            X, WvT[layer], bv + layer * Dm, Vb, Mrows, Dm, Dm, 0);

        qk_sample_kernel<<<(Bc * Hn * Lseq) / 256, blk, 0, stream>>>(Qb, Kb, idx_l, Msc);
        topk_kernel<<<Bc * Hn, 32, 0, stream>>>(Msc, idxtop);
        vmean_kernel<<<Bc * Hn, Eh, 0, stream>>>(Vb, vmean);
        attn_top_kernel<<<Bc * Hn * SK, blk, 0, stream>>>(Qb, Kb, Vb, idxtop, ctxtop);
        assemble_kernel<<<Mrows, blk, 0, stream>>>(vmean, ctxtop, idxtop, XATT);

        gemm_wmma_kernel<<<dim3(Dm / 128, Mrows / 64), blk, 0, stream>>>(
            XATT, WoT[layer], bo + layer * Dm, Sb, Mrows, Dm, Dm, 0);
        layernorm_kernel<<<Mrows, blk, 0, stream>>>(X, Sb, g1 + layer * Dm, beta1 + layer * Dm);

        gemm_wmma_kernel<<<dim3(DFF / 128, Mrows / 64), blk, 0, stream>>>(
            X, W1T[layer], bf1 + layer * DFF, Hb, Mrows, DFF, Dm, 1);
        gemm_wmma_kernel<<<dim3(Dm / 128, Mrows / 64), blk, 0, stream>>>(
            Hb, W2T[layer], bf2 + layer * Dm, Sb, Mrows, Dm, DFF, 0);
        layernorm_kernel<<<Mrows, blk, 0, stream>>>(X, Sb, g2 + layer * Dm, beta2 + layer * Dm);
    }

    layernorm_kernel<<<Mrows, blk, 0, stream>>>(X, nullptr, gf, betaf);
    pool_kernel<<<(Bc * Dm) / 256, blk, 0, stream>>>(X, pooled);
    classify_kernel<<<1, 64, 0, stream>>>(pooled, Wc, bc, out);
}
